// Model_48601849921959
// MI455X (gfx1250) — matrix operand order repaired and checked by reading, not compiled, not run on hardware
//
#include <hip/hip_runtime.h>
#include <math.h>

typedef __attribute__((ext_vector_type(16))) __bf16 v16bf;
typedef __attribute__((ext_vector_type(8)))  float  v8f;

#define B_  16
#define N_  256
#define D_  512
#define H_  8
#define HD  64

#define GF_GELU 1
#define GF_BIAS 2
#define GF_RES  4

__device__ __forceinline__ unsigned int bfu(float f) {
  unsigned u = __builtin_bit_cast(unsigned, f);
  u += 0x7FFFu + ((u >> 16) & 1u);              // round-to-nearest-even
  return u >> 16;
}
__device__ __forceinline__ unsigned int pk2(float lo, float hi) {
  return bfu(lo) | (bfu(hi) << 16);
}
__device__ __forceinline__ float softplusf(float x) {
  return (x > 20.f) ? x : log1pf(expf(x));
}

// ---------------------------------------------------------------------------
// One-time fp32 -> bf16 weight conversion into WMMA *fragment order*:
// out[((kb*(N/16)+tn)*512) + lane*16 + slot], slot i of lane (half,n) holds
// W[kb*32 + half*16 + i][tn*16 + (lane&15)].  Each thread emits one 16B chunk.
// ---------------------------------------------------------------------------
__global__ __launch_bounds__(256)
void cvt_wfrag_kernel(const float* __restrict__ W, unsigned short* __restrict__ out,
                      int K, int N) {
  const int c = blockIdx.x * 256 + threadIdx.x;       // 8-ushort chunk id
  const int total = (K * N) >> 3;
  if (c >= total) return;
  const int blk  = c >> 6;                            // 512-elem tile-block
  const int rem  = c & 63;
  const int lane = rem >> 1;
  const int hs   = (rem & 1) * 8;                     // slot base (0 or 8)
  const int nT   = N >> 4;
  const int kb   = blk / nT;
  const int tn   = blk - kb * nT;
  const int k    = kb * 32 + (lane >> 4) * 16 + hs;
  const int n    = tn * 16 + (lane & 15);
  const float* s = W + (size_t)k * N + n;
  uint4 p;
  p.x = pk2(s[0],             s[(size_t)N]);
  p.y = pk2(s[(size_t)2 * N], s[(size_t)3 * N]);
  p.z = pk2(s[(size_t)4 * N], s[(size_t)5 * N]);
  p.w = pk2(s[(size_t)6 * N], s[(size_t)7 * N]);
  *(uint4*)(out + ((size_t)c << 3)) = p;
}

// ---------------------------------------------------------------------------
// Row RMSNorm, optional bf16 shadow copy of the output (GEMM A operand).
// ---------------------------------------------------------------------------
__global__ __launch_bounds__(256)
void rms_kernel(const float* __restrict__ x, const float* __restrict__ w,
                float* __restrict__ y, unsigned short* __restrict__ ybf,
                int L, float eps, int wrows) {
  __shared__ float red[256];
  const int row = blockIdx.x;
  const int tid = threadIdx.x;
  const float* xr = x + (size_t)row * L;
  float s = 0.f;
  for (int i = tid; i < L; i += 256) { float v = xr[i]; s += v * v; }
  red[tid] = s;
  __syncthreads();
  for (int off = 128; off > 0; off >>= 1) {
    if (tid < off) red[tid] += red[tid + off];
    __syncthreads();
  }
  const float scale = rsqrtf(red[0] / (float)L + eps);
  const float* wr = w + (size_t)(row % wrows) * L;
  for (int i = tid; i < L; i += 256) {
    const float v = xr[i] * scale * wr[i];
    y[(size_t)row * L + i] = v;
    if (ybf) ybf[(size_t)row * L + i] = (unsigned short)bfu(v);
  }
}

// ---------------------------------------------------------------------------
// bf16 WMMA GEMM, LDS-free A, async-copied fragment-ordered B, double buffer.
// 256 threads = 8 waves; block tile 64(M) x 128(N); wave (wm=w&3, wn=w>>2)
// computes a 16x64 strip = 4 WMMAs reusing one A fragment. K step 32.
//  A: bf16 row-major [M][K]; lane fragment = 2x global b128 of one row.
//  Wf: bf16 fragment-order; per k-step the block's 8 tiles are one
//      contiguous 8KB run -> 2x global_load_async_to_lds_b128 per thread.
// ---------------------------------------------------------------------------
__global__ __launch_bounds__(256)
void gemm_bf16_wmma(const unsigned short* __restrict__ A,
                    const unsigned short* __restrict__ Wf,
                    const float* __restrict__ bias,
                    const float* __restrict__ res,
                    float* __restrict__ C,
                    unsigned short* __restrict__ Cbf,
                    int K, int Nout, int flags) {
  __shared__ __align__(32) unsigned int lB[2][2048];   // 2 x 8 KB
  const int tid  = threadIdx.x;
  const int wave = tid >> 5, lane = tid & 31;
  const int half = lane >> 4, l16 = lane & 15;
  const int wm = wave & 3,  wn = wave >> 2;
  const int m0 = blockIdx.y * 64, n0 = blockIdx.x * 128;
  const int nT = Nout >> 4;
  const int steps = K >> 5;

  const unsigned short* wsrc = Wf + (size_t)(n0 >> 4) * 512 + tid * 16;
  const size_t wstep = (size_t)nT * 512;               // elems per k-step

  // prologue: stage step 0 into buffer 0 (32B per thread, dst+src offset:16)
  {
    const unsigned long long g = (unsigned long long)(size_t)wsrc;
    const unsigned d = (unsigned)(size_t)&lB[0][tid * 8];
    asm volatile("global_load_async_to_lds_b128 %0, %1, off\n\t"
                 "global_load_async_to_lds_b128 %0, %1, off offset:16"
                 :: "v"(d), "v"(g) : "memory");
  }

  const unsigned short* Ap = A + (size_t)(m0 + wm * 16 + l16) * K + half * 16;
  v8f acc0 = {}, acc1 = {}, acc2 = {}, acc3 = {};

  for (int s = 0; s < steps; ++s) {
    const int p = s & 1;
    asm volatile("s_wait_asynccnt 0x0" ::: "memory");
    __syncthreads();                                   // buf p ready, buf p^1 free
    if (s + 1 < steps) {
      const unsigned long long g =
          (unsigned long long)(size_t)(wsrc + (size_t)(s + 1) * wstep);
      const unsigned d = (unsigned)(size_t)&lB[p ^ 1][tid * 8];
      asm volatile("global_load_async_to_lds_b128 %0, %1, off\n\t"
                   "global_load_async_to_lds_b128 %0, %1, off offset:16"
                   :: "v"(d), "v"(g) : "memory");
    }
    // A fragment: slots 0..15 = K half*16..half*16+15
    union { uint4 q[2]; v16bf v; } ua;
    ua.q[0] = *(const uint4*)(Ap);
    ua.q[1] = *(const uint4*)(Ap + 8);
    Ap += 32;
    const v16bf* Bf = (const v16bf*)&lB[p][0] + (wn * 4) * 32 + lane;
    acc0 = __builtin_amdgcn_wmma_f32_16x16x32_bf16(false, ua.v, false, Bf[0],
                                                   (short)0, acc0, false, false);
    acc1 = __builtin_amdgcn_wmma_f32_16x16x32_bf16(false, ua.v, false, Bf[32],
                                                   (short)0, acc1, false, false);
    acc2 = __builtin_amdgcn_wmma_f32_16x16x32_bf16(false, ua.v, false, Bf[64],
                                                   (short)0, acc2, false, false);
    acc3 = __builtin_amdgcn_wmma_f32_16x16x32_bf16(false, ua.v, false, Bf[96],
                                                   (short)0, acc3, false, false);
  }

  // epilogue: C/D layout: VGPR r, lane half -> row r + 8*half, col l16
  v8f accs[4] = {acc0, acc1, acc2, acc3};
#pragma unroll
  for (int j = 0; j < 4; ++j) {
    const int gn = n0 + (wn * 4 + j) * 16 + l16;
#pragma unroll
    for (int r = 0; r < 8; ++r) {
      const int gm = m0 + wm * 16 + r + half * 8;
      float val = accs[j][r];
      if (flags & GF_BIAS) val += bias[gn];
      if (flags & GF_GELU) val = 0.5f * val * (1.0f + erff(val * 0.70710678118654752f));
      if (flags & GF_RES)  val += res[(size_t)gm * Nout + gn];
      if (C)   C[(size_t)gm * Nout + gn] = val;
      if (Cbf) Cbf[(size_t)gm * Nout + gn] = (unsigned short)bfu(val);
    }
  }
}

// ---------------------------------------------------------------------------
// S = (tril(LU,-1)+I) @ (triu(LU,1)+diag(softplus(diag)))   [H,64,64]
// ---------------------------------------------------------------------------
__global__ __launch_bounds__(256)
void smat_kernel(const float* __restrict__ LU, float* __restrict__ S) {
  const int h = blockIdx.x;
  const float* M = LU + h * HD * HD;
  for (int t = threadIdx.x; t < HD * HD; t += 256) {
    const int i = t >> 6, j = t & 63;
    const int kmax = (i < j) ? i : j;
    float acc = 0.f;
    for (int k = 0; k <= kmax; ++k) {
      const float l = (k < i) ? M[i * HD + k] : 1.0f;
      const float u = (k < j) ? M[k * HD + j] : softplusf(M[k * HD + k]);
      acc += l * u;
    }
    S[h * HD * HD + t] = acc;
  }
}

__global__ void zero_kernel(float* __restrict__ p, int n) {
  const int i = blockIdx.x * blockDim.x + threadIdx.x;
  if (i < n) p[i] = 0.f;
}

// ---------------------------------------------------------------------------
// Causal linear attention scan for one (b,h): state S[d][e] register-resident.
// Thread (e = tid&63, dg = tid>>6) owns S[dg*16 .. dg*16+16)[e].
// ---------------------------------------------------------------------------
__global__ __launch_bounds__(256)
void attn_kernel(const float* __restrict__ kin, const float* __restrict__ q,
                 const float* __restrict__ v, float* __restrict__ kout,
                 float* __restrict__ ares) {
  const int b = blockIdx.x >> 3;
  const int h = blockIdx.x & 7;
  const int tid = threadIdx.x;
  const int e  = tid & 63;
  const int dg = tid >> 6;
  float S[16];
#pragma unroll
  for (int j = 0; j < 16; ++j) S[j] = 0.f;
  __shared__ float sK[64], sQ[64], sV[64], part[256];
  for (int l = 0; l < N_; ++l) {
    const size_t base = (size_t)(b * N_ + l) * D_ + h * HD;
    if (tid < 64)       sK[tid]       = kin[base + tid];
    else if (tid < 128) sQ[tid - 64]  = q[base + (tid - 64)];
    else if (tid < 192) sV[tid - 128] = v[base + (tid - 128)];
    __syncthreads();
    const float vv = sV[e];
    float p = 0.f;
#pragma unroll
    for (int j = 0; j < 16; ++j) {
      const int dd = dg * 16 + j;
      S[j] += sK[dd] * vv;        // inclusive cumsum of K (x) V
      p    += sQ[dd] * S[j];      // partial Q . S over owned d range
    }
    part[tid] = p;
    __syncthreads();
    if (dg == 0) {
      const float o = part[e] + part[e + 64] + part[e + 128] + part[e + 192];
      kout[base + e] = o;
      if (l & 1)
        ares[(size_t)(b * (N_ / 2) + (l >> 1)) * D_ + h * HD + e] += o;
    }
    __syncthreads();
  }
}

// ---------------------------------------------------------------------------
// out[b,l2,:] = x_orig[b,2*l2+1,:] + per-head attn_res @ S
// ---------------------------------------------------------------------------
__global__ __launch_bounds__(512)
void final_kernel(const float* __restrict__ xorig, const float* __restrict__ ares,
                  const float* __restrict__ Smat, float* __restrict__ out) {
  const int row = blockIdx.x;              // b*128 + l2
  const int b = row >> 7, l2 = row & 127;
  const int tid = threadIdx.x;             // 512
  __shared__ float ar[D_];
  ar[tid] = ares[(size_t)row * D_ + tid];
  __syncthreads();
  const int h = tid >> 6, e = tid & 63;
  const float* Sh  = Smat + h * HD * HD;
  const float* arh = ar + h * HD;
  float acc = xorig[(size_t)(b * N_ + 2 * l2 + 1) * D_ + tid];
#pragma unroll 8
  for (int d0 = 0; d0 < HD; ++d0) acc += arh[d0] * Sh[d0 * HD + e];
  out[(size_t)row * D_ + tid] = acc;
}

// ---------------------------------------------------------------------------
extern "C" void kernel_launch(void* const* d_in, const int* in_sizes, int n_in,
                              void* d_out, int out_size, void* d_ws, size_t ws_size,
                              hipStream_t stream) {
  (void)in_sizes; (void)n_in; (void)out_size; (void)ws_size;
  const float* x     = (const float*)d_in[0];
  const float* ln_w  = (const float*)d_in[1];
  const float* fc_w  = (const float*)d_in[2];
  const float* fc2_w = (const float*)d_in[3];
  const float* fc2_b = (const float*)d_in[4];
  const float* var_w = (const float*)d_in[5];
  const float* q_w   = (const float*)d_in[6];
  const float* v_w   = (const float*)d_in[7];
  const float* qn_w  = (const float*)d_in[8];
  const float* vn_w  = (const float*)d_in[9];
  const float* LU    = (const float*)d_in[10];
  float* out = (float*)d_out;

  const size_t TOK = (size_t)B_ * N_;   // 4096 tokens

  // ---- workspace carve-up: fp32 region then bf16(ushort) region ----
  float* ws    = (float*)d_ws;
  float* xbuf  = ws;  ws += TOK * D_;
  float* t0    = ws;  ws += TOK * D_;
  float* xorig = ws;  ws += TOK * D_;
  float* qbuf  = ws;  ws += TOK * D_;
  float* vbuf  = ws;  ws += TOK * D_;
  float* ares  = ws;  ws += (size_t)B_ * (N_ / 2) * D_;
  float* Smat  = ws;  ws += (size_t)H_ * HD * HD;
  unsigned short* us = (unsigned short*)ws;
  unsigned short* wf1  = us;  us += (size_t)2 * D_ * 4 * D_;   // fc_w frag
  unsigned short* wf2  = us;  us += (size_t)2 * 4 * D_ * D_;   // fc2_w frag
  unsigned short* wqf  = us;  us += (size_t)2 * D_ * D_;       // q_w frag
  unsigned short* wvf  = us;  us += (size_t)2 * D_ * D_;       // v_w frag
  unsigned short* t0bf = us;  us += TOK * D_;                  // rms out bf16
  unsigned short* hbf  = us;  us += TOK * 4 * D_;              // gelu out bf16
  unsigned short* xobf = us;  us += TOK * D_;                  // x_orig bf16

  hipMemcpyAsync(xbuf, x, TOK * D_ * sizeof(float),
                 hipMemcpyDeviceToDevice, stream);

  // ---- one-time weight conversion to bf16 fragment order ----
  for (int i = 0; i < 2; ++i) {
    cvt_wfrag_kernel<<<(D_ * 4 * D_ / 8) / 256, 256, 0, stream>>>(
        fc_w + (size_t)i * D_ * 4 * D_, wf1 + (size_t)i * D_ * 4 * D_, D_, 4 * D_);
    cvt_wfrag_kernel<<<(4 * D_ * D_ / 8) / 256, 256, 0, stream>>>(
        fc2_w + (size_t)i * 4 * D_ * D_, wf2 + (size_t)i * 4 * D_ * D_, 4 * D_, D_);
    cvt_wfrag_kernel<<<(D_ * D_ / 8) / 256, 256, 0, stream>>>(
        q_w + (size_t)i * D_ * D_, wqf + (size_t)i * D_ * D_, D_, D_);
    cvt_wfrag_kernel<<<(D_ * D_ / 8) / 256, 256, 0, stream>>>(
        v_w + (size_t)i * D_ * D_, wvf + (size_t)i * D_ * D_, D_, D_);
  }

  // ---- pre-MLP residual stack ----
  for (int i = 0; i < 2; ++i) {
    rms_kernel<<<(unsigned)TOK, 256, 0, stream>>>(
        xbuf, ln_w + i * D_, t0, t0bf, D_, 1e-5f, 1);
    gemm_bf16_wmma<<<dim3(4 * D_ / 128, (unsigned)TOK / 64), 256, 0, stream>>>(
        t0bf, wf1 + (size_t)i * D_ * 4 * D_, nullptr, nullptr,
        nullptr, hbf, D_, 4 * D_, GF_GELU);
    gemm_bf16_wmma<<<dim3(D_ / 128, (unsigned)TOK / 64), 256, 0, stream>>>(
        hbf, wf2 + (size_t)i * 4 * D_ * D_, fc2_b + i * D_, xbuf,
        xbuf, nullptr, 4 * D_, D_, GF_BIAS | GF_RES);
  }

  rms_kernel<<<(unsigned)TOK, 256, 0, stream>>>(
      xbuf, var_w, xorig, xobf, D_, 1e-5f, 1);
  smat_kernel<<<H_, 256, 0, stream>>>(LU, Smat);
  zero_kernel<<<(B_ * (N_ / 2) * D_) / 256, 256, 0, stream>>>(
      ares, B_ * (N_ / 2) * D_);

  // ---- stacked linear attention ----
  for (int i = 0; i < 2; ++i) {
    gemm_bf16_wmma<<<dim3(D_ / 128, (unsigned)TOK / 64), 256, 0, stream>>>(
        xobf, wqf + (size_t)i * D_ * D_, nullptr, nullptr,
        t0, nullptr, D_, D_, 0);
    rms_kernel<<<(unsigned)(TOK * H_), 256, 0, stream>>>(
        t0, qn_w + i * H_ * HD, qbuf, nullptr, HD, 1e-8f, H_);
    gemm_bf16_wmma<<<dim3(D_ / 128, (unsigned)TOK / 64), 256, 0, stream>>>(
        xobf, wvf + (size_t)i * D_ * D_, nullptr, nullptr,
        t0, nullptr, D_, D_, 0);
    rms_kernel<<<(unsigned)(TOK * H_), 256, 0, stream>>>(
        t0, vn_w + i * H_ * HD, vbuf, nullptr, HD, 1e-8f, H_);
    attn_kernel<<<B_ * H_, 256, 0, stream>>>(
        (i == 0) ? xorig : xbuf, qbuf, vbuf, xbuf, ares);
  }

  final_kernel<<<B_ * (N_ / 2), 512, 0, stream>>>(xorig, ares, Smat, out);
}
